// MQF_75024488726656
// MI455X (gfx1250) — matrix operand order repaired and checked by reading, not compiled, not run on hardware
//
#include <hip/hip_runtime.h>

typedef __attribute__((ext_vector_type(16))) __bf16         v16bf;
typedef __attribute__((ext_vector_type(8)))  float          v8f;
typedef __attribute__((ext_vector_type(16))) unsigned short us16;
typedef __attribute__((ext_vector_type(8)))  unsigned short us8;

#define WMMA_BF16(A, B, C) \
  __builtin_amdgcn_wmma_f32_16x16x32_bf16(false, (A), false, (B), (short)0, (C), false, false)

// CDNA5 async global->LDS copy (ASYNCcnt-tracked, bypasses VGPRs).
#define ASYNC_COPY_B128(ldsaddr_u32, gptr)                           \
  asm volatile("global_load_async_to_lds_b128 %0, %1, off" ::"v"(    \
                   (unsigned)(ldsaddr_u32)),                         \
               "v"(gptr)                                             \
               : "memory")
#define ASYNC_WAIT() asm volatile("s_wait_asynccnt 0x0" ::: "memory")

static __device__ __forceinline__ unsigned short f2bf(float f) {
  unsigned u = __float_as_uint(f);
  return (unsigned short)((u + 0x7fffu + ((u >> 16) & 1u)) >> 16);  // RNE
}

// A-fragment (16-bit, 16x32): per lane two contiguous 8-element K-chunks.
static __device__ __forceinline__ v16bf frag_combine8(const unsigned short* p0,
                                                      const unsigned short* p1) {
  us8 a = *(const us8*)p0;
  us8 b = *(const us8*)p1;
  us16 r;
#pragma unroll
  for (int i = 0; i < 8; ++i) { r[i] = a[i]; r[i + 8] = b[i]; }
  return __builtin_bit_cast(v16bf, r);
}

// B-fragment (16-bit, 32x16): per lane 16 contiguous K values for one column.
static __device__ __forceinline__ v16bf frag_load16(const unsigned short* p) {
  return __builtin_bit_cast(v16bf, *(const us16*)p);
}

static __device__ __forceinline__ v8f vzero8() {
  v8f z;
#pragma unroll
  for (int i = 0; i < 8; ++i) z[i] = 0.0f;
  return z;
}

// ---------------------------------------------------------------------------
// pos_t[h][n][d] = rel_h[h][d][0][hc] + rel_w[h][d][w][0],  n = w*32 + hc
// ---------------------------------------------------------------------------
__global__ __launch_bounds__(256) void pos_build_kernel(
    const float* __restrict__ rel_h, const float* __restrict__ rel_w,
    unsigned short* __restrict__ posd) {
  int id = blockIdx.x * 256 + threadIdx.x;  // h*65536 + n*64 + d
  int d  = id & 63;
  int n  = (id >> 6) & 1023;
  int h  = id >> 16;
  int w  = n >> 5, hc = n & 31;
  float v = rel_h[(h * 64 + d) * 32 + hc] + rel_w[(h * 64 + d) * 32 + w];
  posd[id] = f2bf(v);
}

// ---------------------------------------------------------------------------
// Projection: dst[b][head][n][dd] = bf16( sum_c W[o][c]*x[b][c][n] + bias[o] )
// with o = head*64+dd.  WG = 256 threads (8 waves), tile 128n x 64o, K=512.
// grid.x = 3 * 4 * 8 * 8 (proj, batch, n-block, o-block/head)
// ---------------------------------------------------------------------------
__global__ __launch_bounds__(256) void proj_kernel(
    const float* __restrict__ x0, const float* __restrict__ x1, const float* __restrict__ x2,
    const float* __restrict__ W0, const float* __restrict__ W1, const float* __restrict__ W2,
    const float* __restrict__ b0, const float* __restrict__ b1, const float* __restrict__ b2,
    unsigned short* __restrict__ qd, unsigned short* __restrict__ kd,
    unsigned short* __restrict__ vd) {
  __shared__ unsigned short Xt[128 * 32];  // [n_local][k]  (X transposed, bf16)
  __shared__ unsigned short Ws[64 * 32];   // [o_local][k]  (W rows, bf16)

  int tid = threadIdx.x;
  int wg  = blockIdx.x;
  int ob  = wg & 7;         // o-block == head
  int nb  = (wg >> 3) & 7;  // n-block of 128
  int bt  = (wg >> 6) & 3;  // batch
  int p   = wg >> 8;        // which projection (0=q,1=k,2=v)

  const float* x  = p == 0 ? x0 : (p == 1 ? x1 : x2);
  const float* W  = p == 0 ? W0 : (p == 1 ? W1 : W2);
  const float* bs = p == 0 ? b0 : (p == 1 ? b1 : b2);
  unsigned short* dst = p == 0 ? qd : (p == 1 ? kd : vd);

  int lane = tid & 31, wave = tid >> 5;
  int hi = lane >> 4, lq = lane & 15;

  v8f acc[4] = {vzero8(), vzero8(), vzero8(), vzero8()};

  for (int kk = 0; kk < 16; ++kk) {
    int k0 = kk * 32;
    __syncthreads();
    {  // stage X^T chunk: 32 k x 128 n
      int k  = tid >> 3;
      int ns = (tid & 7) * 16;
      const float* src = x + ((size_t)bt * 512 + k0 + k) * 1024 + nb * 128 + ns;
#pragma unroll
      for (int i = 0; i < 16; ++i) Xt[(ns + i) * 32 + k] = f2bf(src[i]);
      if (kk + 1 < 16)  // pull next chunk toward the WGP (global_prefetch_b8)
        __builtin_prefetch(src + 32 * 1024, 0, 0);
    }
    {  // stage W chunk: 64 o x 32 k (rows contiguous -> B-fragment ready)
      int o  = tid >> 2;
      int ks = (tid & 3) * 8;
      const float* src = W + (size_t)(ob * 64 + o) * 512 + k0 + ks;
      us8 pk;
#pragma unroll
      for (int i = 0; i < 8; ++i) pk[i] = f2bf(src[i]);
      *(us8*)&Ws[o * 32 + ks] = pk;
      if (kk + 1 < 16) __builtin_prefetch(src + 32, 0, 0);
    }
    __syncthreads();

    const unsigned short* xr = &Xt[(wave * 16 + lq) * 32];
    v16bf a = frag_combine8(xr + 16 * hi, xr + 8 + 16 * hi);
#pragma unroll
    for (int t = 0; t < 4; ++t) {
      v16bf b = frag_load16(&Ws[(16 * t + lq) * 32 + 16 * hi]);
      acc[t]  = WMMA_BF16(a, b, acc[t]);
    }
  }

  // epilogue: bias add, convert, store transposed [b][h][n][d]
#pragma unroll
  for (int t = 0; t < 4; ++t) {
    int dd   = 16 * t + lq;
    float bv = bs[ob * 64 + dd];
    unsigned short* drow =
        dst + ((size_t)(bt * 8 + ob) * 1024 + nb * 128 + wave * 16) * 64;
#pragma unroll
    for (int j = 0; j < 8; ++j) {
      int n_loc = 8 * hi + j;  // C-layout row
      drow[(size_t)n_loc * 64 + dd] = f2bf(acc[t][j] + bv);
    }
  }
}

// ---------------------------------------------------------------------------
// Attention staging helpers
// ---------------------------------------------------------------------------
static __device__ __forceinline__ void keff_stage_async(
    const unsigned short* Abh, const unsigned short* Bbh, int m0,
    unsigned short* kbuf, int tid) {
  int m = tid >> 1, sel = tid & 1;
  const unsigned short* src = (sel ? Abh : Bbh) + (size_t)(m0 + m) * 64;
  unsigned dst = (unsigned)(size_t)(void*)&kbuf[m * 128 + sel * 64];
#pragma unroll
  for (int i = 0; i < 8; ++i) ASYNC_COPY_B128(dst + i * 16, src + i * 8);
}

static __device__ __forceinline__ void v_load(const unsigned short* Vbh, int m0,
                                              int tid, uint4 vr[4]) {
  int m = tid >> 1, dc = (tid & 1) * 32;
  const uint4* s4 = (const uint4*)(Vbh + (size_t)(m0 + m) * 64 + dc);
#pragma unroll
  for (int i = 0; i < 4; ++i) vr[i] = s4[i];
}

static __device__ __forceinline__ void vt_store(unsigned short* Vt, int tid,
                                                const uint4 vr[4]) {
  int m = tid >> 1, dc = (tid & 1) * 32;
#pragma unroll
  for (int i = 0; i < 4; ++i) {
    uint4 pk = vr[i];
    int db = dc + i * 8;
    Vt[(db + 0) * 64 + m] = (unsigned short)(pk.x & 0xffff);
    Vt[(db + 1) * 64 + m] = (unsigned short)(pk.x >> 16);
    Vt[(db + 2) * 64 + m] = (unsigned short)(pk.y & 0xffff);
    Vt[(db + 3) * 64 + m] = (unsigned short)(pk.y >> 16);
    Vt[(db + 4) * 64 + m] = (unsigned short)(pk.z & 0xffff);
    Vt[(db + 5) * 64 + m] = (unsigned short)(pk.z >> 16);
    Vt[(db + 6) * 64 + m] = (unsigned short)(pk.w & 0xffff);
    Vt[(db + 7) * 64 + m] = (unsigned short)(pk.w >> 16);
  }
}

// ---------------------------------------------------------------------------
// Fused flash attention, d_eff = 128 (=[a|pos]), d_v = 64.
// WG = 128 threads (4 waves); each wave owns 16 query rows; 64-token K/V tiles
// double-buffered: Keff staged by async-DMA (ASYNCcnt), V transposed via regs.
// grid.x = 3 * 4 * 8 * 16 (branch, batch, head, 64-row block)
// ---------------------------------------------------------------------------
__global__ __launch_bounds__(128) void attn_kernel(
    const unsigned short* __restrict__ qd, const unsigned short* __restrict__ kd,
    const unsigned short* __restrict__ vd, const unsigned short* __restrict__ posd,
    float* __restrict__ out) {
  __shared__ unsigned short Keff[2][64 * 128];  // [m][0:64]=B row, [64:128]=A row
  __shared__ unsigned short Vt[2][64 * 64];     // V transposed: [d][m]
  __shared__ unsigned short Pw[4 * 16 * 64];    // per-wave P staging (C->A relayout)

  int tid  = threadIdx.x;
  int wg   = blockIdx.x;
  int rblk = wg & 15;
  int head = (wg >> 4) & 7;
  int bt   = (wg >> 7) & 3;
  int br   = wg >> 9;  // branch 0/1/2

  const unsigned short* Ap = br == 0 ? qd : (br == 1 ? kd : vd);
  const unsigned short* Bp = br == 0 ? kd : (br == 1 ? vd : qd);
  const unsigned short* Vp = br == 0 ? vd : (br == 1 ? qd : kd);

  size_t bh = ((size_t)bt * 8 + head) * 1024 * 64;
  const unsigned short* Abh = Ap + bh;
  const unsigned short* Bbh = Bp + bh;
  const unsigned short* Vbh = Vp + bh;
  const unsigned short* Pos = posd + (size_t)head * 1024 * 64;

  int lane = tid & 31, wave = tid >> 5;
  int hi = lane >> 4, lq = lane & 15;

  // Qeff fragments: 4 K-chunks of 32 (dims 0..63 from A, 64..127 from pos)
  int nq = rblk * 64 + wave * 16 + lq;  // A-fragment row = lane&15
  v16bf qf[4];
  {
    const unsigned short* ar = Abh + (size_t)nq * 64;
    const unsigned short* pr = Pos + (size_t)nq * 64;
    qf[0] = frag_combine8(ar + 16 * hi,      ar + 8 + 16 * hi);
    qf[1] = frag_combine8(ar + 32 + 16 * hi, ar + 40 + 16 * hi);
    qf[2] = frag_combine8(pr + 16 * hi,      pr + 8 + 16 * hi);
    qf[3] = frag_combine8(pr + 32 + 16 * hi, pr + 40 + 16 * hi);
  }

  float mrun[8], lrun[8];
  v8f O[4];
#pragma unroll
  for (int j = 0; j < 8; ++j) { mrun[j] = -1e30f; lrun[j] = 0.0f; }
#pragma unroll
  for (int t = 0; t < 4; ++t) O[t] = vzero8();

  unsigned short* Pl = &Pw[wave * 1024];
  uint4 vr[4];

  // prologue: stage tile 0
  keff_stage_async(Abh, Bbh, 0, Keff[0], tid);
  v_load(Vbh, 0, tid, vr);
  vt_store(Vt[0], tid, vr);
  ASYNC_WAIT();
  __syncthreads();

  for (int mt = 0; mt < 16; ++mt) {
    const unsigned short* Kc = Keff[mt & 1];
    const unsigned short* Vc = Vt[mt & 1];

    // kick off next tile's DMA before this tile's math (overlap)
    if (mt < 15) {
      keff_stage_async(Abh, Bbh, (mt + 1) * 64, Keff[(mt + 1) & 1], tid);
      v_load(Vbh, (mt + 1) * 64, tid, vr);
    }

    // S = Qeff @ Keff^T  (16 x 64 per wave, K=128 via 4 chained WMMAs)
    v8f S[4];
#pragma unroll
    for (int t = 0; t < 4; ++t) {
      v8f a = vzero8();
      const unsigned short* kr = &Kc[(t * 16 + lq) * 128];
#pragma unroll
      for (int kc = 0; kc < 4; ++kc)
        a = WMMA_BF16(qf[kc], frag_load16(kr + kc * 32 + 16 * hi), a);
      S[t] = a;
    }

    // online softmax: row = 8*hi + j lives in reg j across 16 lanes of the half
#pragma unroll
    for (int j = 0; j < 8; ++j) {
      float mx = fmaxf(fmaxf(S[0][j], S[1][j]), fmaxf(S[2][j], S[3][j]));
#pragma unroll
      for (int sh = 1; sh < 16; sh <<= 1) mx = fmaxf(mx, __shfl_xor(mx, sh, 32));
      float mnew  = fmaxf(mrun[j], mx);
      float scale = __expf(mrun[j] - mnew);
      float ssum  = 0.0f;
#pragma unroll
      for (int t = 0; t < 4; ++t) {
        float e = __expf(S[t][j] - mnew);
        ssum += e;
        Pl[(8 * hi + j) * 64 + 16 * t + lq] = f2bf(e);  // C-layout -> row-major P
      }
#pragma unroll
      for (int sh = 1; sh < 16; sh <<= 1) ssum += __shfl_xor(ssum, sh, 32);
      lrun[j] = lrun[j] * scale + ssum;
      mrun[j] = mnew;
#pragma unroll
      for (int t = 0; t < 4; ++t) O[t][j] *= scale;
    }

    // O += P @ V  (A-fragments of P from wave-private LDS, B from Vt)
    const unsigned short* prw = &Pl[lq * 64];
    v16bf pa0 = frag_combine8(prw + 16 * hi,      prw + 8 + 16 * hi);
    v16bf pa1 = frag_combine8(prw + 32 + 16 * hi, prw + 40 + 16 * hi);
#pragma unroll
    for (int t = 0; t < 4; ++t) {
      const unsigned short* vrow = &Vc[(16 * t + lq) * 64];
      O[t] = WMMA_BF16(pa0, frag_load16(vrow + 16 * hi), O[t]);
      O[t] = WMMA_BF16(pa1, frag_load16(vrow + 32 + 16 * hi), O[t]);
    }

    // finish staging next tile, then publish to all waves
    if (mt < 15) vt_store(Vt[(mt + 1) & 1], tid, vr);
    ASYNC_WAIT();
    __syncthreads();
  }

  // out[(br*4+bt)][head*64+dd][n] = O[n][dd] / l[n]
  size_t obase = (((size_t)br * 4 + bt) * 512 + head * 64) * 1024;
  float invl[8];
#pragma unroll
  for (int j = 0; j < 8; ++j) invl[j] = 1.0f / lrun[j];
#pragma unroll
  for (int t = 0; t < 4; ++t) {
    int dd = 16 * t + lq;
#pragma unroll
    for (int j = 0; j < 8; ++j) {
      int n = rblk * 64 + wave * 16 + 8 * hi + j;
      out[obase + (size_t)dd * 1024 + n] = O[t][j] * invl[j];
    }
  }
}

// ---------------------------------------------------------------------------
extern "C" void kernel_launch(void* const* d_in, const int* in_sizes, int n_in,
                              void* d_out, int out_size, void* d_ws, size_t ws_size,
                              hipStream_t stream) {
  const float* q1    = (const float*)d_in[0];
  const float* q2    = (const float*)d_in[1];
  const float* q3    = (const float*)d_in[2];
  const float* Wq    = (const float*)d_in[3];
  const float* bq    = (const float*)d_in[4];
  const float* Wk    = (const float*)d_in[5];
  const float* bk    = (const float*)d_in[6];
  const float* Wv    = (const float*)d_in[7];
  const float* bv    = (const float*)d_in[8];
  const float* rel_h = (const float*)d_in[9];
  const float* rel_w = (const float*)d_in[10];

  const size_t PROJ_ELEMS = (size_t)4 * 8 * 1024 * 64;  // b,h,n,d
  unsigned short* qd   = (unsigned short*)d_ws;
  unsigned short* kd   = qd + PROJ_ELEMS;
  unsigned short* vd   = kd + PROJ_ELEMS;
  unsigned short* posd = vd + PROJ_ELEMS;  // 8*1024*64

  pos_build_kernel<<<dim3(2048), dim3(256), 0, stream>>>(rel_h, rel_w, posd);
  proj_kernel<<<dim3(768), dim3(256), 0, stream>>>(q1, q2, q3, Wq, Wk, Wv,
                                                   bq, bk, bv, qd, kd, vd);
  attn_kernel<<<dim3(1536), dim3(128), 0, stream>>>(qd, kd, vd, posd,
                                                    (float*)d_out);
}
